// BasicBlock_85684597555721
// MI455X (gfx1250) — matrix operand order repaired and checked by reading, not compiled, not run on hardware
//
#include <hip/hip_runtime.h>
#include <hip/hip_bf16.h>
#include <stdint.h>

typedef __bf16 bf16_t;
typedef __attribute__((ext_vector_type(8)))  __bf16 v8bf;
typedef __attribute__((ext_vector_type(16))) __bf16 v16bf;
typedef __attribute__((ext_vector_type(8)))  float  v8f;

#define C_CH 256
#define HWSZ 3136                 // 56*56
#define NBATCH 32
#define MTOT (NBATCH * HWSZ)      // 100352 output positions
#define WELEMS 589824             // 256*256*9 per conv
#define PW 58                     // padded spatial dim
#define PROW (PW * C_CH)          // 14848 elems per padded row
#define PIMG (PW * PW * C_CH)     // 861184 elems per padded image
#define PTOT ((size_t)NBATCH * PIMG)

// ---------------------------------------------------------------------------
// Fold eval-mode BN into per-channel scale/bias: s = g*rsqrt(v+eps), t = b-m*s
// ---------------------------------------------------------------------------
__global__ void bn_fold_kernel(const float* __restrict__ g, const float* __restrict__ b,
                               const float* __restrict__ m, const float* __restrict__ v,
                               float* __restrict__ s, float* __restrict__ t) {
  int i = threadIdx.x;
  if (i < C_CH) {
    float sc = g[i] * rsqrtf(v[i] + 1e-5f);
    s[i] = sc;
    t[i] = b[i] - m[i] * sc;
  }
}

// ---------------------------------------------------------------------------
// Combine 21 experts (W = sum_e alpha_e * w_e) into bf16, written directly in
// the V_WMMA_F32_16X16X32_BF16 B-fragment layout:
//   wfrag[((tap*8 + cin_chunk)*16 + cout_tile16)*512 + lane*16 + h]
//   lane L: column n = L%16, k = (L>=16 ? 8 : 0) + (h&7) + (h>=8 ? 16 : 0)
// ---------------------------------------------------------------------------
__global__ void combine_weights_kernel(const float* __restrict__ alpha,
                                       const float* __restrict__ w,
                                       bf16_t* __restrict__ wfrag) {
  int idx = blockIdx.x * 256 + threadIdx.x;   // exactly WELEMS threads
  int h     = idx & 15;
  int lane  = (idx >> 4) & 31;
  int ntile = (idx >> 9) & 15;
  int chunk = (idx >> 13) & 7;
  int tap   = idx >> 16;                       // 0..8
  int k    = ((lane >> 4) << 3) + (h & 7) + ((h >> 3) << 4);
  int n    = lane & 15;
  int cout = ntile * 16 + n;
  int cin  = chunk * 32 + k;
  unsigned base = ((unsigned)cout * 256u + (unsigned)cin) * 9u + (unsigned)tap;
  float acc = 0.f;
#pragma unroll
  for (int e = 0; e < 21; ++e)
    acc += alpha[e] * w[(size_t)e * (size_t)WELEMS + base];
  wfrag[idx] = (bf16_t)acc;
}

// ---------------------------------------------------------------------------
// LDS-tiled transpose: NCHW f32 -> padded NHWC bf16 interior.
// Block handles 64 channels x 64 positions. Reads coalesced over hw,
// writes 128B-contiguous chunks over c.  grid = (32*49, 4), block 256.
// ---------------------------------------------------------------------------
__global__ __launch_bounds__(256) void nchw_to_padded_nhwc(const float* __restrict__ x,
                                                           bf16_t* __restrict__ xp) {
  __shared__ float tile[64][65];
  int n   = blockIdx.x / 49;
  int hwt = blockIdx.x % 49;
  int ct  = blockIdx.y;
  int hw0 = hwt * 64;
  int c0  = ct * 64;
#pragma unroll
  for (int i = 0; i < 16; ++i) {
    int idx = i * 256 + threadIdx.x;
    int c = idx >> 6;
    int p = idx & 63;
    tile[c][p] = x[((size_t)n * C_CH + c0 + c) * HWSZ + hw0 + p];
  }
  __syncthreads();
#pragma unroll
  for (int i = 0; i < 16; ++i) {
    int idx = i * 256 + threadIdx.x;
    int p = idx >> 6;
    int c = idx & 63;
    int hw = hw0 + p;
    int h = hw / 56, w = hw % 56;
    xp[(size_t)n * PIMG + (size_t)(h + 1) * PROW + (w + 1) * C_CH + c0 + c] =
        (bf16_t)tile[c][p];
  }
}

// ---------------------------------------------------------------------------
// Zero the 1-wide border of a padded NHWC buffer (228 border cells per image,
// 256 channels each). grid = 912 blocks * 256 threads, 8 bf16 per thread.
// ---------------------------------------------------------------------------
__global__ void zero_border_kernel(bf16_t* __restrict__ buf) {
  int tid = blockIdx.x * 256 + threadIdx.x;   // 233472 total
  int cell  = tid >> 5;
  int cpart = tid & 31;
  int n = cell / 228;
  int b = cell % 228;
  int hh, ww;
  if (b < 58)        { hh = 0;            ww = b;        }
  else if (b < 116)  { hh = 57;           ww = b - 58;   }
  else if (b < 172)  { hh = b - 116 + 1;  ww = 0;        }
  else               { hh = b - 172 + 1;  ww = 57;       }
  v8bf z = {};
  *(v8bf*)(buf + (size_t)n * PIMG + (size_t)hh * PROW + ww * C_CH + cpart * 8) = z;
}

// ---------------------------------------------------------------------------
// Implicit-GEMM 3x3 conv (pad via pre-padded input) with bf16 WMMA.
// Block: 256 threads = 8 waves (4 M-waves x 2 N-waves),
// workgroup tile 256 M x 128 couts; per-wave 64 M x 64 N = 16 accumulators.
// K loop: 9 taps x 8 chunks of 32 cin; per k-step: 8 A-loads + 8 B-loads
// issued up-front (one clause), then 16 v_wmma with partial-count waits.
// FINAL=false: out = bf16 padded-NHWC interior of relu(acc*s + t)
// FINAL=true : out = f32 NCHW of relu(acc*s + t + skip_x)
// ---------------------------------------------------------------------------
template <bool FINAL>
__global__ __launch_bounds__(256) void conv3x3_bf16_wmma(
    const bf16_t* __restrict__ in_pad,
    const bf16_t* __restrict__ wfrag,
    const float*  __restrict__ scale,
    const float*  __restrict__ bias,
    const float*  __restrict__ skip_nchw,
    bf16_t* __restrict__ out_pad,
    float*  __restrict__ out_nchw) {
  const int lane = threadIdx.x & 31;
  const int wid  = threadIdx.x >> 5;
  const int mw = wid & 3;                       // 4 M sub-waves
  const int nw = wid >> 2;                      // 2 N halves
  const int m_base  = (blockIdx.x * 4 + mw) * 64;
  const int nt_base = blockIdx.y * 8 + nw * 4;  // cout tile (units of 16)
  const int ks = (lane & 16) ? 8 : 0;           // A-frag K sub-offset per lane

  // Per-fragment A row base pointers (interior of padded buffer)
  const bf16_t* ap[4];
#pragma unroll
  for (int f = 0; f < 4; ++f) {
    int row = m_base + f * 16 + (lane & 15);
    int img = row / HWSZ;
    int hw  = row % HWSZ;
    int h = hw / 56, w = hw % 56;
    ap[f] = in_pad + (size_t)img * PIMG + (size_t)(h + 1) * PROW + (w + 1) * C_CH + ks;
  }
  const bf16_t* wbase = wfrag + (size_t)nt_base * 512 + lane * 16;

  v8f acc[4][4] = {};

  for (int dh = -1; dh <= 1; ++dh) {
    for (int dw = -1; dw <= 1; ++dw) {
      const int tap  = (dh + 1) * 3 + (dw + 1);
      const int toff = (dh * PW + dw) * C_CH;            // element offset in padded buf
      const bf16_t* wpt = wbase + (size_t)tap * 65536;   // 8 chunks * 16 tiles * 512
#pragma unroll
      for (int chunk = 0; chunk < 8; ++chunk) {
        // Preload all 4 A fragments (one clause of 8 b128 loads)
        v16bf a[4];
#pragma unroll
        for (int f = 0; f < 4; ++f) {
          const bf16_t* rp = ap[f] + toff + chunk * 32;
          v8bf lo = *(const v8bf*)(rp);        // K c0+ks    .. +7
          v8bf hi = *(const v8bf*)(rp + 16);   // K c0+ks+16 .. +23
          a[f] = __builtin_shufflevector(lo, hi, 0,1,2,3,4,5,6,7,
                                                 8,9,10,11,12,13,14,15);
        }
        // B fragments for the 4 cout tiles (L2/WGP$-resident)
        v16bf b0 = *(const v16bf*)(wpt + chunk * 8192 + 0 * 512);
        v16bf b1 = *(const v16bf*)(wpt + chunk * 8192 + 1 * 512);
        v16bf b2 = *(const v16bf*)(wpt + chunk * 8192 + 2 * 512);
        v16bf b3 = *(const v16bf*)(wpt + chunk * 8192 + 3 * 512);
#pragma unroll
        for (int f = 0; f < 4; ++f) {
          acc[f][0] = __builtin_amdgcn_wmma_f32_16x16x32_bf16(
              false, a[f], false, b0, (short)0, acc[f][0], false, false);
          acc[f][1] = __builtin_amdgcn_wmma_f32_16x16x32_bf16(
              false, a[f], false, b1, (short)0, acc[f][1], false, false);
          acc[f][2] = __builtin_amdgcn_wmma_f32_16x16x32_bf16(
              false, a[f], false, b2, (short)0, acc[f][2], false, false);
          acc[f][3] = __builtin_amdgcn_wmma_f32_16x16x32_bf16(
              false, a[f], false, b3, (short)0, acc[f][3], false, false);
        }
      }
    }
  }

  // Epilogue. C/D 16x16 f32 layout: lane<16 -> N=lane, M=r; lane>=16 -> N=lane-16, M=8+r
  const int ncol  = lane & 15;
  const int m_off = (lane >> 4) << 3;
#pragma unroll
  for (int f = 0; f < 4; ++f) {
#pragma unroll
    for (int r = 0; r < 8; ++r) {
      const int m  = m_base + f * 16 + m_off + r;
      const int im = m / HWSZ;
      const int ph = m % HWSZ;
      const int h  = ph / 56, w = ph % 56;
#pragma unroll
      for (int nt = 0; nt < 4; ++nt) {
        const int cout = (nt_base + nt) * 16 + ncol;
        float vv = acc[f][nt][r] * scale[cout] + bias[cout];
        if (FINAL) {
          const size_t o = ((size_t)im * C_CH + cout) * HWSZ + ph;
          vv += skip_nchw[o];
          out_nchw[o] = vv > 0.f ? vv : 0.f;
        } else {
          vv = vv > 0.f ? vv : 0.f;
          out_pad[(size_t)im * PIMG + (size_t)(h + 1) * PROW + (w + 1) * C_CH + cout] =
              (bf16_t)vv;
        }
      }
    }
  }
}

// ---------------------------------------------------------------------------
extern "C" void kernel_launch(void* const* d_in, const int* in_sizes, int n_in,
                              void* d_out, int out_size, void* d_ws, size_t ws_size,
                              hipStream_t stream) {
  const float* x     = (const float*)d_in[0];
  const float* alpha = (const float*)d_in[1];
  const float* w1    = (const float*)d_in[2];
  const float* w2    = (const float*)d_in[3];
  const float* g1 = (const float*)d_in[4];
  const float* b1 = (const float*)d_in[5];
  const float* m1 = (const float*)d_in[6];
  const float* v1 = (const float*)d_in[7];
  const float* g2 = (const float*)d_in[8];
  const float* b2 = (const float*)d_in[9];
  const float* m2 = (const float*)d_in[10];
  const float* v2 = (const float*)d_in[11];

  uint8_t* ws = (uint8_t*)d_ws;
  auto carve = [&](size_t bytes) -> void* {
    void* p = (void*)ws;
    ws += (bytes + 255) & ~(size_t)255;
    return p;
  };
  bf16_t* wf1 = (bf16_t*)carve((size_t)WELEMS * sizeof(bf16_t));
  bf16_t* wf2 = (bf16_t*)carve((size_t)WELEMS * sizeof(bf16_t));
  bf16_t* xp  = (bf16_t*)carve(PTOT * sizeof(bf16_t));
  bf16_t* y1p = (bf16_t*)carve(PTOT * sizeof(bf16_t));
  float* s1 = (float*)carve(C_CH * sizeof(float));
  float* t1 = (float*)carve(C_CH * sizeof(float));
  float* s2 = (float*)carve(C_CH * sizeof(float));
  float* t2 = (float*)carve(C_CH * sizeof(float));

  bn_fold_kernel<<<1, 256, 0, stream>>>(g1, b1, m1, v1, s1, t1);
  bn_fold_kernel<<<1, 256, 0, stream>>>(g2, b2, m2, v2, s2, t2);

  combine_weights_kernel<<<WELEMS / 256, 256, 0, stream>>>(alpha, w1, wf1);
  combine_weights_kernel<<<WELEMS / 256, 256, 0, stream>>>(alpha, w2, wf2);

  zero_border_kernel<<<912, 256, 0, stream>>>(xp);
  zero_border_kernel<<<912, 256, 0, stream>>>(y1p);

  nchw_to_padded_nhwc<<<dim3(NBATCH * 49, 4), 256, 0, stream>>>(x, xp);

  dim3 grid(MTOT / 256, 2);
  conv3x3_bf16_wmma<false><<<grid, 256, 0, stream>>>(xp,  wf1, s1, t1, nullptr, y1p, nullptr);
  conv3x3_bf16_wmma<true ><<<grid, 256, 0, stream>>>(y1p, wf2, s2, t2, x, nullptr, (float*)d_out);
}
